// MaskedAttention_2491081031765
// MI455X (gfx1250) — matrix operand order repaired and checked by reading, not compiled, not run on hardware
//
#include <hip/hip_runtime.h>

typedef __attribute__((ext_vector_type(16))) __bf16 v16bf;
typedef __attribute__((ext_vector_type(8)))  float  v8f;

#define BB 4
#define TT 2048
#define CC 1024
#define NH 16
#define HD 64

union BF16Frag { v16bf v; unsigned int u[8]; };
union F32Frag  { v8f v; float f[8]; };

__device__ __forceinline__ unsigned short f2bf(float f) {
    __bf16 h = (__bf16)f;                       // native f32->bf16 cvt (RNE)
    return __builtin_bit_cast(unsigned short, h);
}
__device__ __forceinline__ unsigned int f2bf2(float lo, float hi) {
#if __has_builtin(__builtin_amdgcn_cvt_pk_bf16_f32)
    auto t = __builtin_amdgcn_cvt_pk_bf16_f32(lo, hi);   // v_cvt_pk_bf16_f32
    return __builtin_bit_cast(unsigned int, t);
#else
    return (unsigned int)f2bf(lo) | ((unsigned int)f2bf(hi) << 16);
#endif
}
// Low 32 bits of a flat shared-memory pointer == LDS byte offset (ISA 10.2).
__device__ __forceinline__ unsigned int lds_off(const void* p) {
    return (unsigned int)(unsigned long long)(uintptr_t)p;
}
// CDNA5 async global->LDS copy, 16B per lane, tracked by ASYNCcnt (GV mode).
__device__ __forceinline__ void async_b128(unsigned int lds, const void* g) {
    unsigned long long a = (unsigned long long)(uintptr_t)g;
    asm volatile("global_load_async_to_lds_b128 %0, %1, off" :: "v"(lds), "v"(a) : "memory");
}
__device__ __forceinline__ void wait_async0() {
    asm volatile("s_wait_asynccnt 0x0" ::: "memory");
}

// Async-stage a 64-row x 128-byte tile (64x64 bf16), row stride in elements.
__device__ __forceinline__ void stage_async_tile(unsigned int ldsbase,
                                                 const unsigned short* g,
                                                 size_t rowstride, int tid) {
#pragma unroll
    for (int i = 0; i < 4; ++i) {
        int idx = tid + i * 128;                  // 512 x 16B chunks
        int rr = idx >> 3, c8 = idx & 7;
        async_b128(ldsbase + idx * 16, g + (size_t)rr * rowstride + c8 * 8);
    }
}
// Stage+convert a 64x64 f32 tile to bf16 LDS (float4 loads, packed dword stores).
__device__ __forceinline__ void stage_conv64(unsigned short* dst, const float* src,
                                             int row0, int k0, int tid) {
#pragma unroll
    for (int i = 0; i < 8; ++i) {
        int idx = tid + i * 128;                  // 0..1023
        int r = idx >> 4, c4 = idx & 15;
        float4 v = *(const float4*)(src + (size_t)(row0 + r) * CC + k0 + c4 * 4);
        unsigned int* d = (unsigned int*)(dst + r * 64 + c4 * 4);
        d[0] = f2bf2(v.x, v.y);
        d[1] = f2bf2(v.z, v.w);
    }
}

// A fragment (16x32 bf16) from LDS row-major [row][k].
__device__ __forceinline__ v16bf load_afrag(const unsigned short* s, int row0, int ld, int k0) {
    int lane = threadIdx.x & 31;
    int h = lane >> 4, r = lane & 15;
    const unsigned short* p = s + (row0 + r) * ld + k0 + 16 * h;
    BF16Frag f;
#pragma unroll
    for (int q = 0; q < 4; ++q) {
        f.u[q]     = *(const unsigned int*)(p + 2 * q);
        f.u[q + 4] = *(const unsigned int*)(p + 8 + 2 * q);
    }
    return f.v;
}
// B fragment (32x16 bf16) from LDS holding B^T row-major: Bt[n][k].
__device__ __forceinline__ v16bf load_bfrag(const unsigned short* s, int col0, int ld, int k0) {
    int lane = threadIdx.x & 31;
    int h = lane >> 4, c = lane & 15;
    const unsigned short* p = s + (col0 + c) * ld + k0 + 16 * h;
    BF16Frag f;
#pragma unroll
    for (int q = 0; q < 8; ++q) f.u[q] = *(const unsigned int*)(p + 2 * q);
    return f.v;
}

#define WMMA_BF16(a, b, c) \
    __builtin_amdgcn_wmma_f32_16x16x32_bf16(false, (a), false, (b), (short)0, (c), false, false)

// ---------------- Kernel 1: QKV projection  y = x @ W^T + b   (double-buffered)
// q,k -> bf16 [B,N,T,D];  v -> bf16 TRANSPOSED per head [B*N, D, T]
__global__ __launch_bounds__(128) void qkv_kernel(
    const float* __restrict__ x,
    const float* __restrict__ Wq, const float* __restrict__ bq,
    const float* __restrict__ Wk, const float* __restrict__ bk,
    const float* __restrict__ Wv, const float* __restrict__ bv,
    unsigned short* __restrict__ qb, unsigned short* __restrict__ kb,
    unsigned short* __restrict__ vb)
{
    __shared__ __align__(16) unsigned short sx[2][64 * 64];
    __shared__ __align__(16) unsigned short sw[2][64 * 64];
    const int tid = threadIdx.x, wid = tid >> 5, lane = tid & 31;
    const int h = lane >> 4, cn = lane & 15;
    const int m0 = blockIdx.x * 64;
    const int n0 = blockIdx.y * 64;
    const float* W  = (blockIdx.z == 0) ? Wq : (blockIdx.z == 1) ? Wk : Wv;
    const float* bs = (blockIdx.z == 0) ? bq : (blockIdx.z == 1) ? bk : bv;

    F32Frag acc[4];
#pragma unroll
    for (int nt = 0; nt < 4; ++nt)
#pragma unroll
        for (int r = 0; r < 8; ++r) acc[nt].f[r] = 0.f;

    stage_conv64(sx[0], x, m0, 0, tid);
    stage_conv64(sw[0], W, n0, 0, tid);
    __syncthreads();

    int cur = 0;
    for (int k0 = 0; k0 < CC; k0 += 64) {
        const int nxt = cur ^ 1;
        if (k0 + 64 < CC) {                        // prefetch next K-step
            stage_conv64(sx[nxt], x, m0, k0 + 64, tid);
            stage_conv64(sw[nxt], W, n0, k0 + 64, tid);
        }
        const unsigned short* cx = sx[cur];
        const unsigned short* cw = sw[cur];
        const int mr = wid * 16;
        v16bf bf[4];
#pragma unroll
        for (int kk = 0; kk < 64; kk += 32) {
            v16bf a = load_afrag(cx, mr, 64, kk);
#pragma unroll
            for (int nt = 0; nt < 4; ++nt) bf[nt] = load_bfrag(cw, nt * 16, 64, kk);
#pragma unroll
            for (int nt = 0; nt < 4; ++nt) acc[nt].v = WMMA_BF16(a, bf[nt], acc[nt].v);
        }
        __syncthreads();
        cur = nxt;
    }

    if (blockIdx.z == 2) {                         // V: transposed per head [bh][d][t]
#pragma unroll
        for (int nt = 0; nt < 4; ++nt) {
            int cg = n0 + nt * 16 + cn;
            float bias = bs[cg];
            int head = cg >> 6, d = cg & 63;
#pragma unroll
            for (int r = 0; r < 8; ++r) {
                int mg = m0 + wid * 16 + r + 8 * h;
                int bidx = mg >> 11, t = mg & (TT - 1);
                vb[(((size_t)bidx * NH + head) * HD + d) * TT + t] = f2bf(acc[nt].f[r] + bias);
            }
        }
    } else {
        unsigned short* out = (blockIdx.z == 0) ? qb : kb;
#pragma unroll
        for (int nt = 0; nt < 4; ++nt) {
            int cg = n0 + nt * 16 + cn;
            float bias = bs[cg];
            int head = cg >> 6, d = cg & 63;
#pragma unroll
            for (int r = 0; r < 8; ++r) {
                int mg = m0 + wid * 16 + r + 8 * h;
                int bidx = mg >> 11, t = mg & (TT - 1);
                out[(((size_t)bidx * NH + head) * TT + t) * HD + d] = f2bf(acc[nt].f[r] + bias);
            }
        }
    }
}

// ---------------- Kernel 2: flash attention, causal, online softmax (double-buffered)
__global__ __launch_bounds__(128) void flash_kernel(
    const unsigned short* __restrict__ qb, const unsigned short* __restrict__ kb,
    const unsigned short* __restrict__ vb /* transposed per head */,
    unsigned short* __restrict__ ob)
{
    __shared__ __align__(16) unsigned short sk[2][64 * 64];
    __shared__ __align__(16) unsigned short sv[2][64 * 64];  // V^T tiles: [d][s]
    __shared__ __align__(16) unsigned short sq[64 * 64];
    __shared__ __align__(16) unsigned short sp[64 * 64];     // per-wave P tiles
    const int tid = threadIdx.x, wid = tid >> 5, lane = tid & 31;
    const int h = lane >> 4, cn = lane & 15;
    const int t0 = blockIdx.x * 64;
    const int bh = blockIdx.y;               // b*NH + n
    const int bb = bh >> 4, hn = bh & 15;
    const unsigned short* qptr  = qb + (size_t)bh * TT * HD;
    const unsigned short* kptr  = kb + (size_t)bh * TT * HD;
    const unsigned short* vtptr = vb + (size_t)bh * HD * TT;

    // prologue: Q tile + first K/V tile, all async
    stage_async_tile(lds_off(sq),    qptr + (size_t)t0 * HD, HD, tid);
    stage_async_tile(lds_off(sk[0]), kptr,                   HD, tid);
    stage_async_tile(lds_off(sv[0]), vtptr,                  TT, tid);
    wait_async0();
    __syncthreads();
    v16bf aq0 = load_afrag(sq, wid * 16, 64, 0);
    v16bf aq1 = load_afrag(sq, wid * 16, 64, 32);

    float m_i[8], l_i[8];
    F32Frag o[4];
#pragma unroll
    for (int r = 0; r < 8; ++r) { m_i[r] = -__builtin_inff(); l_i[r] = 0.f; }
#pragma unroll
    for (int nt = 0; nt < 4; ++nt)
#pragma unroll
        for (int r = 0; r < 8; ++r) o[nt].f[r] = 0.f;

    unsigned short* spw = sp + wid * 16 * 64;

    int cur = 0;
    for (int s0 = 0; s0 <= t0; s0 += 64) {
        const int nxt = cur ^ 1;
        if (s0 + 64 <= t0) {                 // async prefetch next key tile
            stage_async_tile(lds_off(sk[nxt]), kptr + (size_t)(s0 + 64) * HD, HD, tid);
            stage_async_tile(lds_off(sv[nxt]), vtptr + (s0 + 64),             TT, tid);
        }
        const unsigned short* skc = sk[cur];
        const unsigned short* svc = sv[cur];

        // S = Q K^T
        F32Frag sc[4];
#pragma unroll
        for (int nt = 0; nt < 4; ++nt)
#pragma unroll
            for (int r = 0; r < 8; ++r) sc[nt].f[r] = 0.f;
        {
            v16bf bf[4];
#pragma unroll
            for (int kk = 0; kk < 64; kk += 32) {
                v16bf a = (kk == 0) ? aq0 : aq1;
#pragma unroll
                for (int nt = 0; nt < 4; ++nt) bf[nt] = load_bfrag(skc, nt * 16, 64, kk);
#pragma unroll
                for (int nt = 0; nt < 4; ++nt) sc[nt].v = WMMA_BF16(a, bf[nt], sc[nt].v);
            }
        }
        // scale + causal mask (C layout: row = r+8h, col = nt*16 + cn)
        const float scale = 0.125f; // 1/sqrt(64)
#pragma unroll
        for (int nt = 0; nt < 4; ++nt) {
            int sg = s0 + nt * 16 + cn;
#pragma unroll
            for (int r = 0; r < 8; ++r) {
                int tg = t0 + wid * 16 + r + 8 * h;
                float vv = sc[nt].f[r] * scale;
                sc[nt].f[r] = (sg <= tg) ? vv : -__builtin_inff();
            }
        }
        // online softmax
        float mnew[8], alpha[8];
#pragma unroll
        for (int r = 0; r < 8; ++r) {
            float mx = fmaxf(fmaxf(sc[0].f[r], sc[1].f[r]), fmaxf(sc[2].f[r], sc[3].f[r]));
            mx = fmaxf(mx, __shfl_xor(mx, 1));
            mx = fmaxf(mx, __shfl_xor(mx, 2));
            mx = fmaxf(mx, __shfl_xor(mx, 4));
            mx = fmaxf(mx, __shfl_xor(mx, 8));
            mnew[r] = fmaxf(m_i[r], mx);
            alpha[r] = __expf(m_i[r] - mnew[r]);
            m_i[r] = mnew[r];
        }
#pragma unroll
        for (int r = 0; r < 8; ++r) {
            float sum = 0.f;
#pragma unroll
            for (int nt = 0; nt < 4; ++nt) {
                float p = __expf(sc[nt].f[r] - mnew[r]);
                sum += p;
                spw[(r + 8 * h) * 64 + nt * 16 + cn] = f2bf(p);
            }
            sum += __shfl_xor(sum, 1);
            sum += __shfl_xor(sum, 2);
            sum += __shfl_xor(sum, 4);
            sum += __shfl_xor(sum, 8);
            l_i[r] = l_i[r] * alpha[r] + sum;
#pragma unroll
            for (int nt = 0; nt < 4; ++nt) o[nt].f[r] *= alpha[r];
        }
        // O += P @ V
        {
            v16bf bf[4];
#pragma unroll
            for (int kk = 0; kk < 64; kk += 32) {
                v16bf ap = load_afrag(spw, 0, 64, kk);
#pragma unroll
                for (int nt = 0; nt < 4; ++nt) bf[nt] = load_bfrag(svc, nt * 16, 64, kk);
#pragma unroll
                for (int nt = 0; nt < 4; ++nt) o[nt].v = WMMA_BF16(ap, bf[nt], o[nt].v);
            }
        }
        wait_async0();        // prefetch landed (overlapped with compute above)
        __syncthreads();
        cur = nxt;
    }
    // normalize + write bf16 [B,T,C]
#pragma unroll
    for (int r = 0; r < 8; ++r) {
        float inv = 1.0f / l_i[r];
        int tg = t0 + wid * 16 + r + 8 * h;
#pragma unroll
        for (int nt = 0; nt < 4; ++nt) {
            int cg = hn * HD + nt * 16 + cn;
            ob[((size_t)bb * TT + tg) * CC + cg] = f2bf(o[nt].f[r] * inv);
        }
    }
}

// ---------------- Kernel 3: output projection  out = o @ Wp^T + bp  (f32 out)
__global__ __launch_bounds__(128) void proj_kernel(
    const unsigned short* __restrict__ ob, const float* __restrict__ Wp,
    const float* __restrict__ bp, float* __restrict__ out)
{
    __shared__ __align__(16) unsigned short sx[2][64 * 64];
    __shared__ __align__(16) unsigned short sw[2][64 * 64];
    const int tid = threadIdx.x, wid = tid >> 5, lane = tid & 31;
    const int h = lane >> 4, cn = lane & 15;
    const int m0 = blockIdx.x * 64;
    const int n0 = blockIdx.y * 64;

    F32Frag acc[4];
#pragma unroll
    for (int nt = 0; nt < 4; ++nt)
#pragma unroll
        for (int r = 0; r < 8; ++r) acc[nt].f[r] = 0.f;

    stage_async_tile(lds_off(sx[0]), ob + (size_t)m0 * CC, CC, tid);
    stage_conv64(sw[0], Wp, n0, 0, tid);
    wait_async0();
    __syncthreads();

    int cur = 0;
    for (int k0 = 0; k0 < CC; k0 += 64) {
        const int nxt = cur ^ 1;
        if (k0 + 64 < CC) {
            stage_async_tile(lds_off(sx[nxt]), ob + (size_t)m0 * CC + k0 + 64, CC, tid);
            stage_conv64(sw[nxt], Wp, n0, k0 + 64, tid);
        }
        const unsigned short* cx = sx[cur];
        const unsigned short* cw = sw[cur];
        const int mr = wid * 16;
        v16bf bf[4];
#pragma unroll
        for (int kk = 0; kk < 64; kk += 32) {
            v16bf a = load_afrag(cx, mr, 64, kk);
#pragma unroll
            for (int nt = 0; nt < 4; ++nt) bf[nt] = load_bfrag(cw, nt * 16, 64, kk);
#pragma unroll
            for (int nt = 0; nt < 4; ++nt) acc[nt].v = WMMA_BF16(a, bf[nt], acc[nt].v);
        }
        wait_async0();
        __syncthreads();
        cur = nxt;
    }
#pragma unroll
    for (int nt = 0; nt < 4; ++nt) {
        int cg = n0 + nt * 16 + cn;
        float bias = bp[cg];
#pragma unroll
        for (int r = 0; r < 8; ++r) {
            int mg = m0 + wid * 16 + r + 8 * h;
            out[(size_t)mg * CC + cg] = acc[nt].f[r] + bias;
        }
    }
}

extern "C" void kernel_launch(void* const* d_in, const int* in_sizes, int n_in,
                              void* d_out, int out_size, void* d_ws, size_t ws_size,
                              hipStream_t stream) {
    const float* x  = (const float*)d_in[0];
    const float* Wq = (const float*)d_in[1];
    const float* bq = (const float*)d_in[2];
    const float* Wk = (const float*)d_in[3];
    const float* bk = (const float*)d_in[4];
    const float* Wv = (const float*)d_in[5];
    const float* bv = (const float*)d_in[6];
    const float* Wp = (const float*)d_in[7];
    const float* bp = (const float*)d_in[8];

    const size_t BTC = (size_t)BB * TT * CC;
    unsigned short* qb = (unsigned short*)d_ws;
    unsigned short* kb = qb + BTC;
    unsigned short* vb = kb + BTC;     // stored transposed per head [B*N, D, T]
    unsigned short* ob = vb + BTC;

    qkv_kernel<<<dim3((BB * TT) / 64, CC / 64, 3), 128, 0, stream>>>(
        x, Wq, bq, Wk, bk, Wv, bv, qb, kb, vb);
    flash_kernel<<<dim3(TT / 64, BB * NH), 128, 0, stream>>>(qb, kb, vb, ob);
    proj_kernel<<<dim3((BB * TT) / 64, CC / 64), 128, 0, stream>>>(ob, Wp, bp, (float*)d_out);
}
